// SupConLoss2_49967649521746
// MI455X (gfx1250) — matrix operand order repaired and checked by reading, not compiled, not run on hardware
//
#include <hip/hip_runtime.h>
#include <hip/hip_bf16.h>

typedef __attribute__((ext_vector_type(16))) _Float16 v16h;
typedef __attribute__((ext_vector_type(8)))  _Float16 v8h;
typedef __attribute__((ext_vector_type(8)))  float    v8f;

#define N_PTS 4096
#define D_DIM 256
#define ROWS  16           // strip rows per workgroup (WMMA M=16)
#define NTHREADS 512       // 16 waves of 32
#define NWAVES 16
#define TEMP_INV (1.0f / 0.07f)

// ---------------- prep: f32 features -> f16 workspace; zero the output ----------------
__global__ void supcon_prep(const float* __restrict__ feat,
                            _Float16* __restrict__ featH,
                            float* __restrict__ out) {
    int i = blockIdx.x * blockDim.x + threadIdx.x;      // covers N*D exactly
    featH[i] = (_Float16)feat[i];
    if (i == 0) out[0] = 0.0f;
}

// ---------------- wave32 reductions ----------------
__device__ __forceinline__ float wave_max(float v) {
    #pragma unroll
    for (int o = 16; o > 0; o >>= 1) v = fmaxf(v, __shfl_xor(v, o, 32));
    return v;
}
__device__ __forceinline__ float wave_sum(float v) {
    #pragma unroll
    for (int o = 16; o > 0; o >>= 1) v += __shfl_xor(v, o, 32);
    return v;
}
__device__ __forceinline__ int wave_isum(int v) {
    #pragma unroll
    for (int o = 16; o > 0; o >>= 1) v += __shfl_xor(v, o, 32);
    return v;
}

// ---------------- fused strip GEMM + masked-softmax loss ----------------
__launch_bounds__(NTHREADS, 1)
__global__ void supcon_main(const int* __restrict__ labels,
                            const _Float16* __restrict__ featH,
                            float* __restrict__ out) {
    __shared__ float    strip[ROWS][N_PTS];   // 256 KB: 16-row strip of sim
    __shared__ _Float16 aTile[ROWS][D_DIM];   // 8 KB:   f16 A tile (rows r0..r0+15)
    __shared__ int      lab[N_PTS];           // 16 KB:  all labels

    const int tid  = threadIdx.x;
    const int r0   = blockIdx.x * ROWS;
    const int lane = tid & 31;
    const int wv   = tid >> 5;
    const int m    = lane & 15;   // row (A/C) or col (B/C) index within tile
    const int hi   = lane >> 4;   // half-wave selector

    // cooperative loads: labels + A tile (already f16 in workspace)
    for (int k = tid; k < N_PTS; k += NTHREADS) lab[k] = labels[k];
    for (int k = tid; k < ROWS * D_DIM; k += NTHREADS)
        (&aTile[0][0])[k] = featH[(size_t)r0 * D_DIM + k];
    __syncthreads();

    // ---- Phase 1: strip = A(16xD) * F^T, one 16x16 tile per wave per step ----
    // A frag (16-bit A 16x32): lanes<16 hold K={kb..kb+7, kb+16..kb+23}, lanes>=16 shift by 8.
    // B frag (16-bit B 32x16): K order as A: lanes<16 hold K={kb..kb+7, kb+16..kb+23}, lanes>=16 shift by 8;
    //                          column n = lane%16 -> feature row 16j+n (B = F^T).
    for (int j = wv; j < N_PTS / 16; j += NWAVES) {
        v8f acc = {};
        #pragma unroll
        for (int kb = 0; kb < D_DIM; kb += 32) {
            const v8h* pa = (const v8h*)&aTile[m][kb + hi * 8];
            v8h a0 = pa[0];                // K = kb + hi*8 + [0..7]
            v8h a1 = pa[2];                // K = kb + 16 + hi*8 + [0..7]
            const v8h* pb = (const v8h*)(featH + (size_t)(j * 16 + m) * D_DIM + kb + hi * 8);
            v8h b0 = pb[0];                // K = kb + hi*8 + [0..7]
            v8h b1 = pb[2];                // K = kb + 16 + hi*8 + [0..7]
            v16h a, b;
            #pragma unroll
            for (int t = 0; t < 8; ++t) {
                a[t] = a0[t]; a[t + 8] = a1[t];
                b[t] = b0[t]; b[t + 8] = b1[t];
            }
            acc = __builtin_amdgcn_wmma_f32_16x16x32_f16(
                false, a, false, b, (short)0, acc, false, false);
        }
        // C/D layout: lane holds col n=m; VGPR v -> row (hi*8 + v)
        #pragma unroll
        for (int v = 0; v < 8; ++v)
            strip[hi * 8 + v][j * 16 + m] = acc[v];
    }
    __syncthreads();

    // ---- Phase 2: wave wv processes strip row wv (global row r) ----
    const int   r      = r0 + wv;
    const int   myLab  = lab[r];
    const float* row   = strip[wv];

    // Pass A: row maxes (all / same-excl-self / diff) + positive count
    float m_all = -1e30f, m_same = -1e30f, m_diff = -1e30f;
    int cnt = 0;
    for (int k = lane; k < N_PTS; k += 32) {
        float s = row[k];
        bool same = (lab[k] == myLab);
        m_all = fmaxf(m_all, s);
        if (same) {
            if (k != r) { m_same = fmaxf(m_same, s); ++cnt; }
        } else {
            m_diff = fmaxf(m_diff, s);
        }
    }
    m_all  = wave_max(m_all);
    m_same = wave_max(m_same);
    m_diff = wave_max(m_diff);
    cnt    = wave_isum(cnt);

    // Pass B: masked softmax denominators
    float e_same = 0.0f, e_diff = 0.0f;
    for (int k = lane; k < N_PTS; k += 32) {
        float s = row[k];
        bool same = (lab[k] == myLab);
        if (same) { if (k != r) e_same += __expf(s - m_same); }
        else                    e_diff += __expf(s - m_diff);
    }
    e_same = wave_sum(e_same);
    e_diff = wave_sum(e_diff);
    const float inv_es = 1.0f / fmaxf(e_same, 1e-30f);
    const float inv_ed = 1.0f / fmaxf(e_diff, 1e-30f);

    // Pass C: weighted LSE denominator S, positive-weighted logit sum T1, weight sum W
    float S = 0.0f, T1 = 0.0f, W = 0.0f;
    for (int k = lane; k < N_PTS; k += 32) {
        float s = row[k];
        bool same = (lab[k] == myLab);
        float l  = (s - m_all) * TEMP_INV;      // stabilized logits
        float el = __expf(l);
        if (same) {
            if (k != r) {
                float sm = __expf(s - m_same) * inv_es;
                sm = fminf(fmaxf(sm, 1e-8f), 1.0f - 1e-8f);
                float wk = 1.0f - sm;           // w_same
                S  += el * wk;
                T1 += l * wk;
                W  += wk;
            }
        } else {
            float sm = __expf(s - m_diff) * inv_ed;
            sm = fminf(fmaxf(sm, 1e-8f), 1.0f - 1e-8f);
            S += el * (1.0f + sm);              // w_diff
        }
    }
    S  = wave_sum(S);
    T1 = wave_sum(T1);
    W  = wave_sum(W);

    if (lane == 0) {
        float logS = __logf(S + 1e-8f);
        float P    = T1 - logS * W;             // sum(log_prob * w_same) over row
        float ms   = (cnt == 0) ? 1.0f : (float)cnt;
        atomicAdd(out, (-P / ms) * (1.0f / (float)N_PTS));
    }
}

extern "C" void kernel_launch(void* const* d_in, const int* in_sizes, int n_in,
                              void* d_out, int out_size, void* d_ws, size_t ws_size,
                              hipStream_t stream) {
    const float* feat   = (const float*)d_in[0];   // [4096, 256] f32
    const int*   labels = (const int*)d_in[1];     // [4096] int
    float*       out    = (float*)d_out;           // scalar loss
    _Float16*    featH  = (_Float16*)d_ws;         // 2 MB f16 copy of features

    // 1) convert features to f16 + zero the accumulator
    supcon_prep<<<(N_PTS * D_DIM) / 256, 256, 0, stream>>>(feat, featH, out);

    // 2) fused strip-GEMM + loss: one 16-row strip per workgroup
    supcon_main<<<N_PTS / ROWS, NTHREADS, 0, stream>>>(labels, featH, out);
}
